// GroupedQueryAttention_54554674593855
// MI455X (gfx1250) — matrix operand order repaired and checked by reading, not compiled, not run on hardware
//
#include <hip/hip_runtime.h>
#include <hip/hip_bf16.h>

typedef __bf16 bf16;
typedef __attribute__((ext_vector_type(16))) __bf16 v16bf;
typedef __attribute__((ext_vector_type(8)))  __bf16 v8bf;
typedef __attribute__((ext_vector_type(8)))  float  v8f;

#define B_   2
#define S_   2048
#define DIN  2048
#define HQ   32
#define HK   8
#define HD   64
#define NTOT 3072   /* 2048 Q cols + 512 K cols + 512 V cols */

union FragBF { v16bf v; v8bf h[2]; };

// A-matrix fragment (16x32 bf16, MxK), row-major source with leading dim ld.
__device__ __forceinline__ v16bf load_a_frag(const bf16* base, int ld, int m, int half, int koff) {
  FragBF f;
  const bf16* p = base + (size_t)m * ld + koff + half * 8;
  f.h[0] = *reinterpret_cast<const v8bf*>(p);
  f.h[1] = *reinterpret_cast<const v8bf*>(p + 16);
  return f.v;
}

// B-matrix fragment (32x16 bf16, KxN) from an N-major (B^T row-major) source.
__device__ __forceinline__ v16bf load_b_frag(const bf16* base, int ld, int n, int half, int koff) {
  FragBF f;
  const bf16* p = base + (size_t)n * ld + koff + half * 8;
  f.h[0] = *reinterpret_cast<const v8bf*>(p);
  f.h[1] = *reinterpret_cast<const v8bf*>(p + 16);
  return f.v;
}

__device__ __forceinline__ float rowmax16(float v) {
  v = fmaxf(v, __shfl_xor(v, 1, 32));
  v = fmaxf(v, __shfl_xor(v, 2, 32));
  v = fmaxf(v, __shfl_xor(v, 4, 32));
  v = fmaxf(v, __shfl_xor(v, 8, 32));
  return v;
}
__device__ __forceinline__ float rowsum16(float v) {
  v += __shfl_xor(v, 1, 32);
  v += __shfl_xor(v, 2, 32);
  v += __shfl_xor(v, 4, 32);
  v += __shfl_xor(v, 8, 32);
  return v;
}

// ---- CDNA5 async global->LDS copy (ASYNCcnt path), inline asm per ISA 08_async_tensor ----
__device__ __forceinline__ uint32_t lds_off(const void* p) {
  // generic LDS pointer = {aperture_hi32, lds_byte_offset32}; low 32 bits are the DS address
  return (uint32_t)(uintptr_t)p;
}
__device__ __forceinline__ void async_copy_b128(uint32_t lds_dst, const void* gsrc) {
  asm volatile("global_load_async_to_lds_b128 %0, %1, off"
               :: "v"(lds_dst), "v"(gsrc) : "memory");
}
__device__ __forceinline__ void wait_async0() {
  asm volatile("s_wait_asynccnt 0x0" ::: "memory");
}

// ---------------- conversion kernels ----------------
__global__ void cvt_x_kernel(const float* __restrict__ x, bf16* __restrict__ xb, int n) {
  int i = blockIdx.x * blockDim.x + threadIdx.x;
  if (i < n) xb[i] = (bf16)x[i];
}

// Build W^T in bf16: Wt[n][k], n in [0,3072): [0,2048)=Wq, [2048,2560)=Wk, [2560,3072)=Wv.
__global__ void cvt_w_kernel(const float* __restrict__ Wq, const float* __restrict__ Wk,
                             const float* __restrict__ Wv, bf16* __restrict__ Wt, int n) {
  int i = blockIdx.x * blockDim.x + threadIdx.x;
  if (i >= n) return;
  int nn = i >> 11;
  int k  = i & (DIN - 1);
  float v;
  if (nn < 2048)      v = Wq[(size_t)k * 2048 + nn];
  else if (nn < 2560) v = Wk[(size_t)k * 512 + (nn - 2048)];
  else                v = Wv[(size_t)k * 512 + (nn - 2560)];
  Wt[i] = (bf16)v;
}

// ---------------- fused QKV GEMM + RoPE + layout kernel ----------------
// Double-buffered LDS, async DMA overlapped with WMMA, one barrier per K-tile.
__global__ __launch_bounds__(256)
void qkv_gemm_rope_kernel(const bf16* __restrict__ xb, const bf16* __restrict__ Wt,
                          bf16* __restrict__ Qb, bf16* __restrict__ Kb, bf16* __restrict__ Vt) {
  __shared__ bf16 As[2][128][40];   // 2 x (128 rows x 32 k, pad->40)
  __shared__ bf16 Bs[2][128][40];

  const int t    = threadIdx.x;
  const int m0   = blockIdx.y * 128;
  const int n0   = blockIdx.x * 128;
  const int wave = t >> 5, lane = t & 31;
  const int half = lane >> 4, lm = lane & 15;
  const int wm   = (wave & 1) * 64;
  const int wn   = (wave >> 1) * 32;

  auto issue_tile = [&](int buf, int k0) {
    #pragma unroll
    for (int i = 0; i < 2; ++i) {
      int task = t + 256 * i;
      int row  = task >> 2;
      int cg   = (task & 3) << 3;
      async_copy_b128(lds_off(&As[buf][row][cg]), &xb[(size_t)(m0 + row) * DIN + k0 + cg]);
      async_copy_b128(lds_off(&Bs[buf][row][cg]), &Wt[(size_t)(n0 + row) * DIN + k0 + cg]);
    }
  };

  v8f acc[4][2];
  #pragma unroll
  for (int im = 0; im < 4; ++im)
    #pragma unroll
    for (int in = 0; in < 2; ++in)
      #pragma unroll
      for (int r = 0; r < 8; ++r) acc[im][in][r] = 0.0f;

  constexpr int NT = DIN / 32;
  issue_tile(0, 0);
  for (int kt = 0; kt < NT; ++kt) {
    const int buf = kt & 1;
    wait_async0();        // my portion of buffer `buf` has landed in LDS
    __syncthreads();      // everyone's portion landed; everyone done reading buf^1
    if (kt + 1 < NT) issue_tile(buf ^ 1, (kt + 1) * 32);   // DMA next tile under the WMMAs

    v16bf af[4], bfr[2];
    #pragma unroll
    for (int im = 0; im < 4; ++im)
      af[im] = load_a_frag(&As[buf][0][0], 40, wm + im * 16 + lm, half, 0);
    #pragma unroll
    for (int in = 0; in < 2; ++in)
      bfr[in] = load_b_frag(&Bs[buf][0][0], 40, wn + in * 16 + lm, half, 0);
    #pragma unroll
    for (int im = 0; im < 4; ++im)
      #pragma unroll
      for (int in = 0; in < 2; ++in)
        acc[im][in] = __builtin_amdgcn_wmma_f32_16x16x32_bf16(
            false, af[im], false, bfr[in], (short)0, acc[im][in], false, false);
  }

  // Epilogue: RoPE + pack. D-frag: row = r + 8*half, col = lane%16.
  #pragma unroll
  for (int im = 0; im < 4; ++im) {
    #pragma unroll
    for (int in = 0; in < 2; ++in) {
      const int gn = n0 + wn + in * 16 + lm;
      const int dh = gn & 63;
      const int fi = dh >> 1;
      const float freq = __expf(-(float)(2 * fi) * (9.210340371976184f / 64.0f)); // 10000^(-2i/64)
      #pragma unroll
      for (int r = 0; r < 8; ++r) {
        const int gm = m0 + wm + im * 16 + r + 8 * half;
        const int bb = gm >> 11;
        const int ss = gm & (S_ - 1);
        float v  = acc[im][in][r];
        float pr = __shfl_xor(v, 1, 32);   // even<->odd head-dim partner is the adjacent lane
        const bool even = ((dh & 1) == 0);
        const float xe = even ? v : pr;
        const float xo = even ? pr : v;
        float sn, cs;
        __sincosf((float)ss * freq, &sn, &cs);
        const float roped = even ? (xe * cs - xo * sn) : (xe * sn + xo * cs);
        if (gn < 2048) {
          const int h = gn >> 6;
          Qb[(((size_t)bb * HQ + h) * S_ + ss) * HD + dh] = (bf16)(roped * 0.125f);
        } else if (gn < 2560) {
          const int h = (gn - 2048) >> 6;
          Kb[(((size_t)bb * HK + h) * S_ + ss) * HD + dh] = (bf16)roped;
        } else {
          const int h = (gn - 2560) >> 6;
          Vt[(((size_t)bb * HK + h) * HD + dh) * S_ + ss] = (bf16)v;  // no RoPE on V
        }
      }
    }
  }
}

// ---------------- causal GQA flash attention ----------------
__global__ __launch_bounds__(256)
void gqa_attn_kernel(const bf16* __restrict__ Qb, const bf16* __restrict__ Kb,
                     const bf16* __restrict__ Vt, float* __restrict__ out) {
  __shared__ bf16 Ks[2][32][72];    // double-buffered: 32 keys x 64 dims
  __shared__ bf16 Vs[2][64][40];    // double-buffered: V^T chunk 64 dims x 32 keys
  __shared__ bf16 Ps[8][16][40];    // per-wave P staging (D-layout -> A-layout)

  const int t    = threadIdx.x;
  const int wave = t >> 5, lane = t & 31, half = lane >> 4, lm = lane & 15;
  const int bh = blockIdx.y;
  const int bb = bh >> 5;
  const int hq = bh & 31;
  const int hk = hq >> 2;
  const int q0 = blockIdx.x * 128;
  const int qw = q0 + wave * 16;

  const bf16* Qbase = Qb + ((size_t)bb * HQ + hq) * S_ * HD;
  const bf16* Kbase = Kb + ((size_t)bb * HK + hk) * S_ * HD;
  const bf16* Vbase = Vt + ((size_t)bb * HK + hk) * HD * S_;

  auto issue_kv = [&](int buf, int j0) {
    int krow = t >> 3, kcg = (t & 7) << 3;
    async_copy_b128(lds_off(&Ks[buf][krow][kcg]), &Kbase[(size_t)(j0 + krow) * HD + kcg]);
    int vrow = t >> 2, vcg = (t & 3) << 3;
    async_copy_b128(lds_off(&Vs[buf][vrow][vcg]), &Vbase[(size_t)vrow * S_ + j0 + vcg]);
  };

  v16bf qf[2];
  #pragma unroll
  for (int kd = 0; kd < 2; ++kd)
    qf[kd] = load_a_frag(Qbase, HD, qw + lm, half, kd * 32);

  v8f o[4];
  float mrow[8], lrow[8];
  #pragma unroll
  for (int r = 0; r < 8; ++r) {
    mrow[r] = -3.0e38f; lrow[r] = 0.0f;
    #pragma unroll
    for (int f = 0; f < 4; ++f) o[f][r] = 0.0f;
  }

  const int nb = q0 / 32 + 4;       // causal: key blocks [0, q0+128)
  issue_kv(0, 0);
  for (int kt = 0; kt < nb; ++kt) {
    const int buf = kt & 1;
    const int j0  = kt * 32;
    wait_async0();
    __syncthreads();
    if (kt + 1 < nb) issue_kv(buf ^ 1, j0 + 32);   // DMA next K/V under the compute

    if (j0 <= qw + 15) {            // wave-uniform causal skip
      v8f sc[2];
      #pragma unroll
      for (int cn = 0; cn < 2; ++cn)
        #pragma unroll
        for (int r = 0; r < 8; ++r) sc[cn][r] = 0.0f;

      #pragma unroll
      for (int kd = 0; kd < 2; ++kd) {
        #pragma unroll
        for (int cn = 0; cn < 2; ++cn) {
          v16bf kf = load_b_frag(&Ks[buf][0][0], 72, cn * 16 + lm, half, kd * 32);
          sc[cn] = __builtin_amdgcn_wmma_f32_16x16x32_bf16(
              false, qf[kd], false, kf, (short)0, sc[cn], false, false);
        }
      }

      #pragma unroll
      for (int r = 0; r < 8; ++r) {
        const int q = qw + r + 8 * half;
        #pragma unroll
        for (int cn = 0; cn < 2; ++cn) {
          const int j = j0 + cn * 16 + lm;
          if (j > q) sc[cn][r] = -1.0e30f;   // causal mask (Q pre-scaled by 1/8)
        }
        float mb = rowmax16(fmaxf(sc[0][r], sc[1][r]));
        const float mnew  = fmaxf(mrow[r], mb);
        const float alpha = __expf(mrow[r] - mnew);
        const float p0 = __expf(sc[0][r] - mnew);
        const float p1 = __expf(sc[1][r] - mnew);
        lrow[r] = lrow[r] * alpha + rowsum16(p0 + p1);
        mrow[r] = mnew;
        #pragma unroll
        for (int f = 0; f < 4; ++f) o[f][r] *= alpha;
        const int prow = r + 8 * half;
        Ps[wave][prow][lm]      = (bf16)p0;
        Ps[wave][prow][16 + lm] = (bf16)p1;
      }
      __threadfence_block();   // order wave-private LDS staging stores before cross-lane reads

      v16bf pf = load_a_frag(&Ps[wave][0][0], 40, lm, half, 0);
      #pragma unroll
      for (int f = 0; f < 4; ++f) {
        v16bf vf = load_b_frag(&Vs[buf][0][0], 40, f * 16 + lm, half, 0);
        o[f] = __builtin_amdgcn_wmma_f32_16x16x32_bf16(
            false, pf, false, vf, (short)0, o[f], false, false);
      }
    }
  }

  #pragma unroll
  for (int r = 0; r < 8; ++r) {
    const int q = qw + r + 8 * half;
    const float inv_l = 1.0f / lrow[r];
    #pragma unroll
    for (int f = 0; f < 4; ++f)
      out[((size_t)bb * S_ + q) * (HQ * HD) + hq * HD + f * 16 + lm] = o[f][r] * inv_l;
  }
}

extern "C" void kernel_launch(void* const* d_in, const int* in_sizes, int n_in,
                              void* d_out, int out_size, void* d_ws, size_t ws_size,
                              hipStream_t stream) {
  const float* x  = (const float*)d_in[0];
  const float* Wq = (const float*)d_in[1];
  const float* Wk = (const float*)d_in[2];
  const float* Wv = (const float*)d_in[3];
  float* out = (float*)d_out;

  char* p = (char*)d_ws;
  bf16* xb = (bf16*)p; p += (size_t)B_ * S_ * DIN * sizeof(bf16);        // 16 MB
  bf16* Wt = (bf16*)p; p += (size_t)NTOT * DIN * sizeof(bf16);           // 12 MB
  bf16* Qb = (bf16*)p; p += (size_t)B_ * HQ * S_ * HD * sizeof(bf16);    // 16 MB
  bf16* Kb = (bf16*)p; p += (size_t)B_ * HK * S_ * HD * sizeof(bf16);    //  4 MB
  bf16* Vt = (bf16*)p;                                                   //  4 MB

  { int n = B_ * S_ * DIN; cvt_x_kernel<<<(n + 255) / 256, 256, 0, stream>>>(x, xb, n); }
  { int n = NTOT * DIN;    cvt_w_kernel<<<(n + 255) / 256, 256, 0, stream>>>(Wq, Wk, Wv, Wt, n); }
  qkv_gemm_rope_kernel<<<dim3(NTOT / 128, (B_ * S_) / 128), 256, 0, stream>>>(xb, Wt, Qb, Kb, Vt);
  gqa_attn_kernel<<<dim3(S_ / 128, B_ * HQ), 256, 0, stream>>>(Qb, Kb, Vt, out);
}
